// GNOT_44942537785560
// MI455X (gfx1250) — matrix operand order repaired and checked by reading, not compiled, not run on hardware
//
#include <hip/hip_runtime.h>
#include <math.h>

// ---------------------------------------------------------------------------
// Types for CDNA5 WMMA (wave32): v_wmma_f32_16x16x32_bf16
// ---------------------------------------------------------------------------
typedef __attribute__((ext_vector_type(16))) __bf16 v16bf;
typedef __attribute__((ext_vector_type(8)))  __bf16 v8bf;
typedef __attribute__((ext_vector_type(8)))  float  v8f;

#define BB     4
#define NN     4096
#define MTOT   (BB * NN)        // 16384 tokens
#define DMODEL 256
#define NHEAD  8
#define HDIM   32
#define NEXP   4
#define SZB    ((size_t)MTOT * DMODEL * sizeof(float))   // 16 MiB

__device__ __forceinline__ float gelu_f(float x) {
    return 0.5f * x * (1.0f + erff(x * 0.70710678118654752440f));
}

// ---------------------------------------------------------------------------
// Weight transpose + fp32 -> bf16 convert:  W[K][N] row-major -> Wt[N][K]
// (so each lane's WMMA B-fragment is two runs of 8 contiguous bf16 = two 16B loads)
// ---------------------------------------------------------------------------
__global__ void wcvt_kernel(const float* __restrict__ src, __bf16* __restrict__ dst,
                            int K, int N) {
    int idx = blockIdx.x * 256 + threadIdx.x;
    if (idx >= K * N) return;
    int n = idx / K, k = idx - n * K;
    dst[idx] = (__bf16)src[(size_t)k * N + n];
}

// ---------------------------------------------------------------------------
// bf16 WMMA GEMM: C[M][N] = epilogue( A[M][K]_f32 @ Wt[N][K]_bf16 + bias )
//   act:  0 = none, 1 = exact GELU
//   R  :  if non-null, C = R + val        (residual fuse)
//   gate: if non-null, C += gate[m*4+gi]*val  (MoE expert accumulate)
// Block tile 128x128, 8 wave32 waves (4x2), wave tile 32x64 (2x4 WMMA accs).
// ---------------------------------------------------------------------------
#define LDSA 40   // 32 + 8 bf16 pad per row

__launch_bounds__(256)
__global__ void gemm_bf16_kernel(const float* __restrict__ A,
                                 const __bf16* __restrict__ Wt,
                                 const float* __restrict__ bias,
                                 float* __restrict__ C,
                                 const float* __restrict__ R,
                                 const float* __restrict__ gate,
                                 int M, int N, int K, int gateIdx, int act) {
    __shared__ __bf16 As[128 * LDSA];
    const int tid  = threadIdx.x;
    const int lane = tid & 31, wave = tid >> 5;
    const int wm = wave & 3, wn = wave >> 2;
    const int m0 = blockIdx.y * 128, n0 = blockIdx.x * 128;
    const int half = lane >> 4, l16 = lane & 15;

    v8f acc[2][4] = {};

    for (int k0 = 0; k0 < K; k0 += 32) {
        __syncthreads();
        // Stage A tile 128x32 fp32 -> bf16 LDS (coalesced float4 reads)
#pragma unroll
        for (int i = 0; i < 4; i++) {
            int c4  = tid + i * 256;        // float4 chunk 0..1023
            int row = c4 >> 3;
            int col = (c4 & 7) << 2;
            float4 f = *(const float4*)(A + (size_t)(m0 + row) * K + k0 + col);
            union { __bf16 h[4]; uint2 u; } pk;
            pk.h[0] = (__bf16)f.x; pk.h[1] = (__bf16)f.y;
            pk.h[2] = (__bf16)f.z; pk.h[3] = (__bf16)f.w;
            *(uint2*)&As[row * LDSA + col] = pk.u;
        }
        __syncthreads();

        // A fragments per ISA 16-bit A layout:
        // lanes 0-15: K[0..7],K[16..23]; lanes 16-31: K[8..15],K[24..31]
        v16bf afrag[2];
#pragma unroll
        for (int mi = 0; mi < 2; mi++) {
            const __bf16* p = &As[(wm * 32 + mi * 16 + l16) * LDSA + half * 8];
            union { v16bf v; v8bf h[2]; } ua;
            ua.h[0] = *(const v8bf*)(p);
            ua.h[1] = *(const v8bf*)(p + 16);
            afrag[mi] = ua.v;
        }

        // B fragments: col-major weights -> K[8h..8h+7], K[16+8h..16+8h+7] per lane
#pragma unroll
        for (int ni = 0; ni < 4; ni++) {
            int n = n0 + wn * 64 + ni * 16 + l16;
            const __bf16* pb = Wt + (size_t)n * K + k0 + half * 8;
            union { v16bf v; v8bf h[2]; } ub;
            ub.h[0] = *(const v8bf*)(pb);
            ub.h[1] = *(const v8bf*)(pb + 16);
            v16bf bfrag = ub.v;
#pragma unroll
            for (int mi = 0; mi < 2; mi++) {
                acc[mi][ni] = __builtin_amdgcn_wmma_f32_16x16x32_bf16(
                    false, afrag[mi], false, bfrag, (short)0, acc[mi][ni],
                    false, false);
            }
        }
    }

    // Epilogue per C/D layout: VGPR r -> M = r + 8*half, N = l16
#pragma unroll
    for (int ni = 0; ni < 4; ni++) {
        int n = n0 + wn * 64 + ni * 16 + l16;
        float bv = bias[n];
#pragma unroll
        for (int mi = 0; mi < 2; mi++) {
#pragma unroll
            for (int r = 0; r < 8; r++) {
                int m = m0 + wm * 32 + mi * 16 + half * 8 + r;
                float v = acc[mi][ni][r] + bv;
                if (act) v = gelu_f(v);
                size_t off = (size_t)m * N + n;
                if (gate)       C[off] += gate[m * NEXP + gateIdx] * v;
                else if (R)     C[off]  = R[off] + v;
                else            C[off]  = v;
            }
        }
    }
}

// ---------------------------------------------------------------------------
// LayerNorm over D=256; one wave32 per row; optional residual fuse.
// ---------------------------------------------------------------------------
__global__ void ln_kernel(const float* __restrict__ x, const float* __restrict__ g,
                          const float* __restrict__ be, const float* __restrict__ res,
                          float* __restrict__ out) {
    int row  = blockIdx.x * 8 + (threadIdx.x >> 5);
    int lane = threadIdx.x & 31;
    const float* xr = x + (size_t)row * DMODEL;
    float v[8]; float s = 0.f;
#pragma unroll
    for (int i = 0; i < 8; i++) { v[i] = xr[lane + i * 32]; s += v[i]; }
    for (int o = 16; o; o >>= 1) s += __shfl_xor(s, o, 32);
    float mean = s * (1.0f / DMODEL);
    float q = 0.f;
#pragma unroll
    for (int i = 0; i < 8; i++) { float d = v[i] - mean; q += d * d; }
    for (int o = 16; o; o >>= 1) q += __shfl_xor(q, o, 32);
    float rstd = rsqrtf(q * (1.0f / DMODEL) + 1e-5f);
#pragma unroll
    for (int i = 0; i < 8; i++) {
        int cc = lane + i * 32;
        float o = (v[i] - mean) * rstd * g[cc] + be[cc];
        size_t oo = (size_t)row * DMODEL + cc;
        out[oo] = res ? res[oo] + o : o;
    }
}

// ---------------------------------------------------------------------------
// Per-head softmax over 32 channels (in place). One wave per (row, head).
// ---------------------------------------------------------------------------
__global__ void softmax32_kernel(float* __restrict__ x) {
    int row = blockIdx.x;
    int h = threadIdx.x >> 5, lane = threadIdx.x & 31;
    size_t off = (size_t)row * DMODEL + h * HDIM + lane;
    float v = x[off];
    float m = v;
    for (int o = 16; o; o >>= 1) m = fmaxf(m, __shfl_xor(m, o, 32));
    float e = __expf(v - m);
    float s = e;
    for (int o = 16; o; o >>= 1) s += __shfl_xor(s, o, 32);
    x[off] = e / s;
}

// ---------------------------------------------------------------------------
// kv[b,h,c,d] = sum_j k[b,j,h,c]*v[b,j,h,d];  ksum[b,h,c] = sum_j k[b,j,h,c]
// One block per (b,h); LDS-tiled over j.
// ---------------------------------------------------------------------------
__global__ void kv_kernel(const float* __restrict__ k, const float* __restrict__ v,
                          float* __restrict__ kvout, float* __restrict__ ksum) {
    __shared__ float ks[128][HDIM];
    __shared__ float vs[128][HDIM];
    int bh = blockIdx.x, b = bh >> 3, h = bh & 7;
    const float* kb = k + (size_t)b * NN * DMODEL + h * HDIM;
    const float* vb = v + (size_t)b * NN * DMODEL + h * HDIM;
    int tid = threadIdx.x, d = tid & 31, c0 = tid >> 5;
    float accv[4] = {0.f, 0.f, 0.f, 0.f};
    float ksacc = 0.f;
    for (int j0 = 0; j0 < NN; j0 += 128) {
        __syncthreads();
#pragma unroll
        for (int i = 0; i < 16; i++) {
            int e = tid + i * 256, jj = e >> 5, cc = e & 31;
            ks[jj][cc] = kb[(size_t)(j0 + jj) * DMODEL + cc];
            vs[jj][cc] = vb[(size_t)(j0 + jj) * DMODEL + cc];
        }
        __syncthreads();
        for (int jj = 0; jj < 128; jj++) {
            float vv = vs[jj][d];
            accv[0] += ks[jj][c0]      * vv;
            accv[1] += ks[jj][c0 + 8]  * vv;
            accv[2] += ks[jj][c0 + 16] * vv;
            accv[3] += ks[jj][c0 + 24] * vv;
        }
        if (tid < 32) for (int jj = 0; jj < 128; jj++) ksacc += ks[jj][tid];
    }
#pragma unroll
    for (int i = 0; i < 4; i++)
        kvout[(size_t)bh * 1024 + (c0 + i * 8) * 32 + d] = accv[i];
    if (tid < 32) ksum[bh * 32 + tid] = ksacc;
}

// ---------------------------------------------------------------------------
// hna apply: out = q + (q @ kv) / (q . ksum). One wave per (row, head).
// ---------------------------------------------------------------------------
__global__ void hna_kernel(const float* __restrict__ q, const float* __restrict__ kv,
                           const float* __restrict__ ksum, float* __restrict__ out) {
    int row = blockIdx.x, b = row >> 12;            // NN = 4096
    int h = threadIdx.x >> 5, lane = threadIdx.x & 31;
    int bh = b * NHEAD + h;
    float qv = q[(size_t)row * DMODEL + h * HDIM + lane];
    float t = qv * ksum[bh * 32 + lane];
    for (int o = 16; o; o >>= 1) t += __shfl_xor(t, o, 32);
    float inv = 1.0f / t;
    const float* kvb = kv + (size_t)bh * 1024;
    float r = 0.f;
    for (int c = 0; c < 32; c++) {
        float qc = __shfl(qv, c, 32);
        r += qc * kvb[c * 32 + lane];
    }
    out[(size_t)row * DMODEL + h * HDIM + lane] = qv + r * inv;
}

// ---------------------------------------------------------------------------
// Tiny-K linear (K<=4), fp32, optional GELU: out[m][n] = act(sum_k A*W + b)
// ---------------------------------------------------------------------------
__global__ void lin_smallK_kernel(const float* __restrict__ A, const float* __restrict__ W,
                                  const float* __restrict__ b, float* __restrict__ out,
                                  int M, int N, int K, int act) {
    int idx = blockIdx.x * 256 + threadIdx.x;
    if (idx >= M * N) return;
    int m = idx / N, n = idx - m * N;
    float s = b[n];
    for (int k = 0; k < K; k++) s += A[(size_t)m * K + k] * W[k * N + n];
    out[idx] = act ? gelu_f(s) : s;
}

// gate head: (M,512) @ (512,4) + b, softmax over 4 experts
__global__ void gate2_kernel(const float* __restrict__ A, const float* __restrict__ W,
                             const float* __restrict__ b, float* __restrict__ gate, int M) {
    int m = blockIdx.x * 256 + threadIdx.x;
    if (m >= M) return;
    float s[4] = {b[0], b[1], b[2], b[3]};
    const float* a = A + (size_t)m * 512;
    for (int k = 0; k < 512; k++) {
        float av = a[k];
        s[0] += av * W[k * 4 + 0]; s[1] += av * W[k * 4 + 1];
        s[2] += av * W[k * 4 + 2]; s[3] += av * W[k * 4 + 3];
    }
    float mx = fmaxf(fmaxf(s[0], s[1]), fmaxf(s[2], s[3]));
    float e0 = __expf(s[0] - mx), e1 = __expf(s[1] - mx);
    float e2 = __expf(s[2] - mx), e3 = __expf(s[3] - mx);
    float inv = 1.0f / (e0 + e1 + e2 + e3);
    gate[m * 4 + 0] = e0 * inv; gate[m * 4 + 1] = e1 * inv;
    gate[m * 4 + 2] = e2 * inv; gate[m * 4 + 3] = e3 * inv;
}

// final head: (M,256) @ (256,1) + b -> d_out
__global__ void lin_final_kernel(const float* __restrict__ A, const float* __restrict__ W,
                                 const float* __restrict__ b, float* __restrict__ out, int M) {
    int m = blockIdx.x * 256 + threadIdx.x;
    if (m >= M) return;
    float s = b[0];
    const float* a = A + (size_t)m * DMODEL;
    for (int k = 0; k < DMODEL; k++) s += a[k] * W[k];
    out[m] = s;
}

// ===========================================================================
// Host orchestration
// ===========================================================================
struct Lin  { const float* W; const float* b; };
struct LNp  { const float* g; const float* b; };
struct LinB { const __bf16* Wt; const float* b; };

struct PW {
    void* const* d; int i;
    const float* f() { return (const float*)d[i++]; }
};

static const __bf16* cvtW(hipStream_t s, char* base, size_t& off,
                          const float* W, int K, int N) {
    __bf16* dst = (__bf16*)(base + off);
    size_t bytes = (size_t)K * N * sizeof(__bf16);
    off += (bytes + 255) & ~(size_t)255;
    int tot = K * N;
    wcvt_kernel<<<(tot + 255) / 256, 256, 0, s>>>(W, dst, K, N);
    return dst;
}

static inline void gemmL(hipStream_t s, const float* A, const LinB& w, float* C,
                         const float* R, const float* gate, int gateIdx,
                         int N, int K, int act) {
    dim3 g(N / 128, MTOT / 128);
    gemm_bf16_kernel<<<g, 256, 0, s>>>(A, w.Wt, w.b, C, R, gate, MTOT, N, K,
                                       gateIdx, act);
}

extern "C" void kernel_launch(void* const* d_in, const int* in_sizes, int n_in,
                              void* d_out, int out_size, void* d_ws, size_t ws_size,
                              hipStream_t stream) {
    (void)in_sizes; (void)n_in; (void)out_size; (void)ws_size;

    // ---- parse params (setup_inputs dict order, depth-first) ----
    PW pw{d_in, 0};
    const float* x_in = pw.f();             // (B, N1, 2)
    const float* y_in = pw.f();             // (B, N2, 3)
    auto rdLin = [&](Lin& L) { L.W = pw.f(); L.b = pw.f(); };
    auto rdLN  = [&](LNp& L) { L.g = pw.f(); L.b = pw.f(); };

    struct MlpP { Lin inp, hid[3], out; } trunk, branch, outm;
    auto rdMlp = [&](MlpP& m) {
        rdLin(m.inp); for (int i = 0; i < 3; i++) rdLin(m.hid[i]); rdLin(m.out);
    };
    rdMlp(trunk); rdMlp(branch); rdMlp(outm);

    struct AttnP { Lin q, k, v, proj; };
    struct MoeP  { const float *W1, *b1, *W2, *b2; };
    struct BlkP  { AttnP self_, cross; LNp ln1, ln2, ln3, ln4, ln5;
                   MoeP moe1, moe2; Lin g0, g1, g2; } blk[2];
    for (int bi = 0; bi < 2; bi++) {
        BlkP& B = blk[bi];
        rdLin(B.self_.q); rdLin(B.self_.k); rdLin(B.self_.v); rdLin(B.self_.proj);
        rdLin(B.cross.q); rdLin(B.cross.k); rdLin(B.cross.v); rdLin(B.cross.proj);
        rdLN(B.ln1); rdLN(B.ln2); rdLN(B.ln3); rdLN(B.ln4); rdLN(B.ln5);
        B.moe1.W1 = pw.f(); B.moe1.b1 = pw.f(); B.moe1.W2 = pw.f(); B.moe1.b2 = pw.f();
        B.moe2.W1 = pw.f(); B.moe2.b1 = pw.f(); B.moe2.W2 = pw.f(); B.moe2.b2 = pw.f();
        rdLin(B.g0); rdLin(B.g1); rdLin(B.g2);
    }

    // ---- workspace bump allocator ----
    char* wp = (char*)d_ws;
    auto alloc = [&](size_t bytes) -> char* {
        char* p = wp; wp += (bytes + 255) & ~(size_t)255; return p;
    };
    char*  w16    = alloc(20u << 20);          // bf16 weight pool
    size_t w16off = 0;
    float* b_xt  = (float*)alloc(SZB);
    float* b_yt  = (float*)alloc(SZB);
    float* b_t0  = (float*)alloc(SZB);
    float* b_t1  = (float*)alloc(SZB);
    float* b_r   = (float*)alloc(SZB);
    float* b_q   = (float*)alloc(SZB);
    float* b_k   = (float*)alloc(SZB);        // contiguous after b_q
    float* b_v   = (float*)alloc(SZB);
    float* b_acc = (float*)alloc(SZB);
    float* b_h   = (float*)alloc(2 * SZB);    // (M, 512)
    float* b_g1  = b_q;                        // alias: gate runs before q/k live
    float* b_gate = (float*)alloc((size_t)MTOT * 4 * sizeof(float));
    float* b_kv   = (float*)alloc((size_t)BB * NHEAD * 1024 * sizeof(float));
    float* b_ksum = (float*)alloc((size_t)BB * NHEAD * 32 * sizeof(float));

    auto mkLinB = [&](const Lin& L, int K, int N) -> LinB {
        return LinB{ cvtW(stream, w16, w16off, L.W, K, N), L.b };
    };

    // ---- convert all WMMA-path weights to bf16 col-major ----
    struct MlpB { LinB hid[3], out; } trunkB, branchB;
    LinB outm_inpB, outm_hidB[3];
    for (int i = 0; i < 3; i++) trunkB.hid[i]  = mkLinB(trunk.hid[i], 256, 256);
    trunkB.out  = mkLinB(trunk.out, 256, 256);
    for (int i = 0; i < 3; i++) branchB.hid[i] = mkLinB(branch.hid[i], 256, 256);
    branchB.out = mkLinB(branch.out, 256, 256);
    outm_inpB = mkLinB(outm.inp, 256, 256);
    for (int i = 0; i < 3; i++) outm_hidB[i] = mkLinB(outm.hid[i], 256, 256);

    struct AttnB { LinB q, k, v, proj; };
    struct MoeB  { LinB W1[4], W2[4]; };
    struct BlkB  { AttnB self_, cross; MoeB moe1, moe2; LinB g1; } blkB[2];
    for (int bi = 0; bi < 2; bi++) {
        BlkP& P = blk[bi]; BlkB& Q = blkB[bi];
        auto mkAttn = [&](const AttnP& a) -> AttnB {
            return AttnB{ mkLinB(a.q, 256, 256), mkLinB(a.k, 256, 256),
                          mkLinB(a.v, 256, 256), mkLinB(a.proj, 256, 256) };
        };
        Q.self_ = mkAttn(P.self_); Q.cross = mkAttn(P.cross);
        for (int e = 0; e < 4; e++) {
            Q.moe1.W1[e] = LinB{ cvtW(stream, w16, w16off, P.moe1.W1 + (size_t)e*256*512, 256, 512), P.moe1.b1 + e*512 };
            Q.moe1.W2[e] = LinB{ cvtW(stream, w16, w16off, P.moe1.W2 + (size_t)e*512*256, 512, 256), P.moe1.b2 + e*256 };
            Q.moe2.W1[e] = LinB{ cvtW(stream, w16, w16off, P.moe2.W1 + (size_t)e*256*512, 256, 512), P.moe2.b1 + e*512 };
            Q.moe2.W2[e] = LinB{ cvtW(stream, w16, w16off, P.moe2.W2 + (size_t)e*512*256, 512, 256), P.moe2.b2 + e*256 };
        }
        Q.g1 = mkLinB(P.g1, 512, 512);
    }

    // ---- helpers ----
    auto run_mlp_in = [&](const float* X, int Kin, const Lin& inp, const MlpB& mb,
                          float* dst) {
        int tot = MTOT * 256;
        lin_smallK_kernel<<<(tot + 255) / 256, 256, 0, stream>>>(
            X, inp.W, inp.b, b_t0, MTOT, 256, Kin, 1);
        float* cur = b_t0; float* alt = b_t1;
        for (int i = 0; i < 3; i++) {
            gemmL(stream, cur, mb.hid[i], alt, cur, nullptr, 0, 256, 256, 1);
            float* t = cur; cur = alt; alt = t;
        }
        gemmL(stream, cur, mb.out, dst, nullptr, nullptr, 0, 256, 256, 0);
    };

    auto run_attn = [&](const AttnB& a, const float* xln, const float* yln,
                        float* tmp, const float* Rres, float* Out) {
        gemmL(stream, xln, a.q, b_q, nullptr, nullptr, 0, 256, 256, 0);
        gemmL(stream, yln, a.k, b_k, nullptr, nullptr, 0, 256, 256, 0);
        gemmL(stream, yln, a.v, b_v, nullptr, nullptr, 0, 256, 256, 0);
        softmax32_kernel<<<MTOT, 256, 0, stream>>>(b_q);
        softmax32_kernel<<<MTOT, 256, 0, stream>>>(b_k);
        kv_kernel<<<BB * NHEAD, 256, 0, stream>>>(b_k, b_v, b_kv, b_ksum);
        hna_kernel<<<MTOT, 256, 0, stream>>>(b_q, b_kv, b_ksum, tmp);
        gemmL(stream, tmp, a.proj, Out, Rres, nullptr, 0, 256, 256, 0);
    };

    auto run_moe = [&](const MoeB& mw, const LNp& lnp, float* streamBuf) {
        hipMemsetAsync(b_acc, 0, SZB, stream);
        for (int e = 0; e < 4; e++) {
            gemmL(stream, streamBuf, mw.W1[e], b_h, nullptr, nullptr, 0, 512, 256, 1);
            gemmL(stream, b_h, mw.W2[e], b_acc, nullptr, b_gate, e, 256, 512, 0);
        }
        // stream = stream + ln(acc)   (in-place residual fuse)
        ln_kernel<<<MTOT / 8, 256, 0, stream>>>(b_acc, lnp.g, lnp.b, streamBuf, streamBuf);
    };

    // ---- phase 1: trunk / branch MLPs ----
    run_mlp_in(x_in, 2, trunk.inp, trunkB, b_xt);    // xt
    run_mlp_in(y_in, 3, branch.inp, branchB, b_yt);  // yt

    // ---- phase 2: transformer blocks ----
    for (int bi = 0; bi < 2; bi++) {
        BlkP& P = blk[bi]; BlkB& Q = blkB[bi];

        // gate = softmax(g2(gelu(g1(gelu(g0(pos))))))   (pos == x_in)
        {
            int tot = MTOT * 512;
            lin_smallK_kernel<<<(tot + 255) / 256, 256, 0, stream>>>(
                x_in, P.g0.W, P.g0.b, b_g1, MTOT, 512, 2, 1);
            gemmL(stream, b_g1, Q.g1, b_h, nullptr, nullptr, 0, 512, 512, 1);
            gate2_kernel<<<(MTOT + 255) / 256, 256, 0, stream>>>(
                b_h, P.g2.W, P.g2.b, b_gate, MTOT);
        }

        // r = x + cross_attn(ln1(x), ln2(y))
        ln_kernel<<<MTOT / 8, 256, 0, stream>>>(b_xt, P.ln1.g, P.ln1.b, nullptr, b_t0);
        ln_kernel<<<MTOT / 8, 256, 0, stream>>>(b_yt, P.ln2.g, P.ln2.b, nullptr, b_t1);
        run_attn(Q.cross, b_t0, b_t1, /*tmp=*/b_t0, /*res=*/b_xt, /*out=*/b_r);

        // r = r + ln3(moe1(r, gate))
        run_moe(Q.moe1, P.ln3, b_r);

        // r = r + self_attn(ln4(r), ln4(r))
        ln_kernel<<<MTOT / 8, 256, 0, stream>>>(b_r, P.ln4.g, P.ln4.b, nullptr, b_t0);
        run_attn(Q.self_, b_t0, b_t0, /*tmp=*/b_t1, /*res=*/b_r, /*out=*/b_xt);

        // r = r + ln5(moe2(r, gate))   (stream now in b_xt)
        run_moe(Q.moe2, P.ln5, b_xt);
    }

    // ---- phase 3: output MLP (256 -> ... -> 1) ----
    gemmL(stream, b_xt, outm_inpB, b_t0, nullptr, nullptr, 0, 256, 256, 1);
    {
        float* cur = b_t0; float* alt = b_t1;
        for (int i = 0; i < 3; i++) {
            gemmL(stream, cur, outm_hidB[i], alt, cur, nullptr, 0, 256, 256, 1);
            float* t = cur; cur = alt; alt = t;
        }
        lin_final_kernel<<<(MTOT + 255) / 256, 256, 0, stream>>>(
            cur, outm.out.W, outm.out.b, (float*)d_out, MTOT);
    }
}
